// Transformer_45320494907483
// MI455X (gfx1250) — matrix operand order repaired and checked by reading, not compiled, not run on hardware
//
#include <hip/hip_runtime.h>
#include <hip/hip_bf16.h>
#include <math.h>

// ---------------------------------------------------------------------------
// Types for CDNA5 WMMA (gfx1250, wave32)
// ---------------------------------------------------------------------------
typedef __attribute__((ext_vector_type(16))) _Float16 v16h;
typedef __attribute__((ext_vector_type(8)))  _Float16 v8h;
typedef __attribute__((ext_vector_type(8)))  float    v8f;
typedef __attribute__((ext_vector_type(4)))  float    f32x4;
typedef __attribute__((ext_vector_type(4)))  unsigned u32x4;
typedef __attribute__((ext_vector_type(8)))  int      i32x8;
typedef __attribute__((ext_vector_type(4)))  int      i32x4;

union U16 { v16h h; unsigned u[8]; };
union HU8 { v8h v; _Float16 f[8]; };

#define DIMC 768
#define NH 12
#define HD 64
#define NSEQ 1024
#define NB 4
#define HID 3072
#define TOK (NB * NSEQ)   // 4096

enum { EPI_F16 = 0, EPI_F32_BIAS_RES = 1, EPI_F16_BIAS_GELU = 2 };

#if defined(__has_builtin)
#  if __has_builtin(__builtin_amdgcn_tensor_load_to_lds) && \
      __has_builtin(__builtin_amdgcn_s_wait_tensorcnt)
#    define USE_TDM 1
#  endif
#endif

__device__ __forceinline__ float gelu_exact(float x) {
    return 0.5f * x * (1.0f + erff(x * 0.70710678118654752f));
}

// ---------------------------------------------------------------------------
// f32 -> f16 weight conversion
// ---------------------------------------------------------------------------
__global__ void f32_to_f16_kernel(const float* __restrict__ in,
                                  _Float16* __restrict__ out, int n) {
    int i = blockIdx.x * 256 + threadIdx.x;
    if (i < n) out[i] = (_Float16)in[i];
}

// ---------------------------------------------------------------------------
// Residual copy via Tensor Data Mover (TDM): global -> LDS (DMA) -> global.
// 16 rows x 768 f32 (48 KB) per block. Falls back to plain copy w/o TDM.
// ---------------------------------------------------------------------------
__global__ __launch_bounds__(256)
void residual_copy_kernel(const float* __restrict__ src, float* __restrict__ dst) {
    __shared__ __attribute__((aligned(16))) float lbuf[16 * DIMC];  // 48 KB
    const int tid = threadIdx.x;
    const int rowBase = blockIdx.x * 16;
#if defined(USE_TDM)
    if (tid < 32) {
        unsigned lds_off = (unsigned)(size_t)(&lbuf[0]);
        unsigned long long ga =
            (unsigned long long)(size_t)(src + (size_t)rowBase * DIMC);
        // D# group0: count=1 | lds_addr | global_addr | type=2
        u32x4 g0 = { 1u,
                     lds_off,
                     (unsigned)(ga & 0xffffffffu),
                     (unsigned)((ga >> 32) & 0x01ffffffu) | (2u << 30) };
        // D# group1: data_size=8B; tensor_dim0=384 (8B units), tensor_dim1=4096;
        // tile_dim0=384, tile_dim1=16; tensor_dim0_stride=384
        i32x8 g1 = { (int)(3u << 16),
                     (int)((384u & 0xffffu) << 16),
                     (int)((4096u & 0xffffu) << 16),
                     (int)(384u << 16),
                     16,
                     384,
                     0, 0 };
        i32x4 gz = { 0, 0, 0, 0 };
#if __clang_major__ >= 23
        i32x8 gx = { 0, 0, 0, 0, 0, 0, 0, 0 };
        __builtin_amdgcn_tensor_load_to_lds(g0, g1, gz, gz, gx, 0);
#else
        __builtin_amdgcn_tensor_load_to_lds(g0, g1, gz, gz, 0);
#endif
        __builtin_amdgcn_s_wait_tensorcnt(0);
    }
    __syncthreads();
    #pragma unroll
    for (int c = 0; c < 12; ++c) {
        int i = tid + c * 256;                               // 3072 float4 chunks
        *(f32x4*)&dst[(size_t)rowBase * DIMC + i * 4] = *(const f32x4*)&lbuf[i * 4];
    }
#else
    #pragma unroll
    for (int c = 0; c < 12; ++c) {
        int i = tid + c * 256;
        *(f32x4*)&dst[(size_t)rowBase * DIMC + i * 4] =
            *(const f32x4*)&src[(size_t)rowBase * DIMC + i * 4];
    }
#endif
}

// ---------------------------------------------------------------------------
// Pair bias: rbf(dist) ++ emb[edge] -> LN(32) -> 32x12 linear -> GELU
// bias layout [B,H,N,N] f32
// ---------------------------------------------------------------------------
__global__ __launch_bounds__(256)
void bias_kernel(const float* __restrict__ dist, const int* __restrict__ edge,
                 const float* __restrict__ emb,
                 const float* __restrict__ lng, const float* __restrict__ lnb,
                 const float* __restrict__ dw, const float* __restrict__ db,
                 float* __restrict__ bias) {
    size_t idx = (size_t)blockIdx.x * 256 + threadIdx.x;   // < 4*1024*1024
    float f[32];
    float d = dist[idx];
    int et = edge[idx];
    const float sigma_inv = 1.0f / 1.25f;                  // sigma = 20/16
    #pragma unroll
    for (int r = 0; r < 16; ++r) {
        float mu = (20.0f / 15.0f) * (float)r;             // linspace(0,20,16)
        float t = (d - mu) * sigma_inv;
        f[r] = __expf(-t * t);
    }
    #pragma unroll
    for (int r = 0; r < 16; ++r) f[16 + r] = emb[et * 16 + r];

    float s = 0.f, s2 = 0.f;
    #pragma unroll
    for (int c = 0; c < 32; ++c) { s += f[c]; s2 += f[c] * f[c]; }
    float mean = s * (1.0f / 32.0f);
    float var = s2 * (1.0f / 32.0f) - mean * mean;
    float rstd = rsqrtf(var + 1e-5f);
    #pragma unroll
    for (int c = 0; c < 32; ++c) f[c] = (f[c] - mean) * rstd * lng[c] + lnb[c];

    size_t bI = idx >> 20;                   // batch
    size_t ij = idx & ((1u << 20) - 1);      // i*1024 + j
    #pragma unroll
    for (int h = 0; h < NH; ++h) {
        float acc = db[h];
        #pragma unroll
        for (int c = 0; c < 32; ++c) acc += dw[h * 32 + c] * f[c];
        bias[((bI * NH + h) << 20) + ij] = gelu_exact(acc);
    }
}

// ---------------------------------------------------------------------------
// Row LayerNorm: [4096, 768] f32 -> f16
// ---------------------------------------------------------------------------
__global__ __launch_bounds__(256)
void ln_kernel(const float* __restrict__ x, const float* __restrict__ g,
               const float* __restrict__ bta, _Float16* __restrict__ out) {
    int row = blockIdx.x;
    int tid = threadIdx.x;
    __shared__ float red[18];
    const float* xr = x + (size_t)row * DIMC;
    float v0 = xr[tid], v1 = xr[tid + 256], v2 = xr[tid + 512];
    float s = v0 + v1 + v2;
    float s2 = v0 * v0 + v1 * v1 + v2 * v2;
    #pragma unroll
    for (int off = 16; off; off >>= 1) {
        s  += __shfl_down(s,  off, 32);
        s2 += __shfl_down(s2, off, 32);
    }
    if ((tid & 31) == 0) { red[tid >> 5] = s; red[8 + (tid >> 5)] = s2; }
    __syncthreads();
    if (tid == 0) {
        float ts = 0.f, ts2 = 0.f;
        #pragma unroll
        for (int i = 0; i < 8; ++i) { ts += red[i]; ts2 += red[8 + i]; }
        float mean = ts * (1.0f / DIMC);
        float var = ts2 * (1.0f / DIMC) - mean * mean;
        red[16] = mean;
        red[17] = rsqrtf(var + 1e-5f);
    }
    __syncthreads();
    float mean = red[16], rstd = red[17];
    _Float16* orow = out + (size_t)row * DIMC;
    orow[tid]       = (_Float16)((v0 - mean) * rstd * g[tid]       + bta[tid]);
    orow[tid + 256] = (_Float16)((v1 - mean) * rstd * g[tid + 256] + bta[tid + 256]);
    orow[tid + 512] = (_Float16)((v2 - mean) * rstd * g[tid + 512] + bta[tid + 512]);
}

// ---------------------------------------------------------------------------
// WMMA GEMM (double-buffered LDS): C[M,N] = A[M,K](f16) @ W[N,K](f16)^T
// 128x128x32 tiles, 256 threads = 8 waves, each wave 64x32 (4x2 WMMA tiles)
// ---------------------------------------------------------------------------
__global__ __launch_bounds__(256)
void gemm_f16_wmma(const _Float16* __restrict__ A, const _Float16* __restrict__ W,
                   void* __restrict__ Cout, const float* __restrict__ bias,
                   const float* __restrict__ res,
                   int M, int N, int K, int mode) {
    __shared__ __attribute__((aligned(16))) _Float16 As[2][128 * 32];  // 2 x 8 KB
    __shared__ __attribute__((aligned(16))) _Float16 Ws[2][128 * 32];  // 2 x 8 KB

    const int tid = threadIdx.x;
    const int lane = tid & 31;
    const int wv = tid >> 5;
    const int wm = wv >> 2;          // 0..1 : 64-row group
    const int wn = wv & 3;           // 0..3 : 32-col group
    const int l15 = lane & 15;
    const int hi = lane >> 4;
    const int rowBase = blockIdx.y * 128;
    const int colBase = blockIdx.x * 128;

    v8f acc[4][2];
    #pragma unroll
    for (int i = 0; i < 4; ++i)
        #pragma unroll
        for (int j = 0; j < 2; ++j) acc[i][j] = {};

    // staging coordinates: thread owns 16B chunks at rows r0 and r0+64
    const int r0 = tid >> 2;
    const int c0 = tid & 3;
    const _Float16* aPtr = A + (size_t)(rowBase + r0) * K + c0 * 8;
    const _Float16* wPtr = W + (size_t)(colBase + r0) * K + c0 * 8;
    const size_t rowOff = (size_t)64 * K;

    u32x4 ra0 = *(const u32x4*)aPtr;
    u32x4 ra1 = *(const u32x4*)(aPtr + rowOff);
    u32x4 rw0 = *(const u32x4*)wPtr;
    u32x4 rw1 = *(const u32x4*)(wPtr + rowOff);

    const int nIter = K >> 5;
    for (int it = 0; it < nIter; ++it) {
        const int buf = it & 1;
        _Float16* Asb = As[buf];
        _Float16* Wsb = Ws[buf];
        *(u32x4*)&Asb[r0 * 32 + c0 * 8] = ra0;
        *(u32x4*)&Asb[(r0 + 64) * 32 + c0 * 8] = ra1;
        *(u32x4*)&Wsb[r0 * 32 + c0 * 8] = rw0;
        *(u32x4*)&Wsb[(r0 + 64) * 32 + c0 * 8] = rw1;

        if (it + 1 < nIter) {
            const _Float16* an = aPtr + (size_t)(it + 1) * 32;
            const _Float16* wn_ = wPtr + (size_t)(it + 1) * 32;
            ra0 = *(const u32x4*)an;
            ra1 = *(const u32x4*)(an + rowOff);
            rw0 = *(const u32x4*)wn_;
            rw1 = *(const u32x4*)(wn_ + rowOff);
            if (it + 2 < nIter) {                 // global_prefetch_b8 path
                __builtin_prefetch(an + 32, 0, 1);
                __builtin_prefetch(wn_ + 32, 0, 1);
            }
        }
        __syncthreads();   // publishes buf; prior compute on buf finished 2 iters ago

        const unsigned* Au = (const unsigned*)As[buf];
        const unsigned* Wu = (const unsigned*)Ws[buf];

        // A fragments: 16x32 f16 layout (lane M = lane&15, k-halves by lane>>4)
        U16 af[4];
        #pragma unroll
        for (int mt = 0; mt < 4; ++mt) {
            int r = wm * 64 + mt * 16 + l15;
            #pragma unroll
            for (int v = 0; v < 8; ++v) {
                int kidx = ((v & 4) ? 16 : 0) + hi * 8 + (v & 3) * 2;
                af[mt].u[v] = Au[r * 16 + (kidx >> 1)];
            }
        }
        // B fragments: 32x16 (K x N); B[k][n] = W[n][k], k pair contiguous
        U16 bf[2];
        #pragma unroll
        for (int nt = 0; nt < 2; ++nt) {
            int n = wn * 32 + nt * 16 + l15;
            #pragma unroll
            for (int v = 0; v < 8; ++v)
                bf[nt].u[v] = Wu[n * 16 + ((v & 4) ? 8 : 0) + hi * 4 + (v & 3)];
        }
        #pragma unroll
        for (int mt = 0; mt < 4; ++mt)
            #pragma unroll
            for (int nt = 0; nt < 2; ++nt)
                acc[mt][nt] = __builtin_amdgcn_wmma_f32_16x16x32_f16(
                    false, af[mt].h, false, bf[nt].h, (short)0, acc[mt][nt],
                    false, false);
    }

    // Epilogue (C layout: row = r + 8*(lane>>4), col = lane&15)
    #pragma unroll
    for (int mt = 0; mt < 4; ++mt) {
        #pragma unroll
        for (int nt = 0; nt < 2; ++nt) {
            #pragma unroll
            for (int r = 0; r < 8; ++r) {
                int row = rowBase + wm * 64 + mt * 16 + r + 8 * hi;
                int col = colBase + wn * 32 + nt * 16 + l15;
                size_t o = (size_t)row * N + col;
                float v = acc[mt][nt][r];
                if (mode == EPI_F16) {
                    ((_Float16*)Cout)[o] = (_Float16)v;
                } else if (mode == EPI_F32_BIAS_RES) {
                    ((float*)Cout)[o] = v + bias[col] + res[o];
                } else { // EPI_F16_BIAS_GELU
                    float sct = v + bias[col];
                    ((_Float16*)Cout)[o] = (_Float16)gelu_exact(sct);
                }
            }
        }
    }
}

// ---------------------------------------------------------------------------
// Flash attention with pair bias. grid = (N/64, B*H), 128 threads (4 waves).
// Each wave owns 16 query rows; K blocks of 64 streamed through LDS.
// bias+mask tiles staged into LDS with b128 loads.
// qkv: [TOK, 3*768] f16 ; out: [TOK, 768] f16
// ---------------------------------------------------------------------------
__global__ __launch_bounds__(128)
void flash_attn_kernel(const _Float16* __restrict__ qkv,
                       const float* __restrict__ pbias,
                       const float* __restrict__ mask,
                       const float* __restrict__ lmask,
                       _Float16* __restrict__ out) {
    const int qb = blockIdx.x;           // 0..15
    const int bh = blockIdx.y;           // 0..47
    const int b  = bh / NH;
    const int h  = bh % NH;
    const int tid = threadIdx.x;
    const int lane = tid & 31;
    const int w = tid >> 5;              // wave 0..3
    const int l15 = lane & 15;
    const int hi = lane >> 4;

    __shared__ __attribute__((aligned(16))) _Float16 Qs[64 * 64];      // 8 KB
    __shared__ __attribute__((aligned(16))) _Float16 Ks[64 * 64];      // 8 KB
    __shared__ __attribute__((aligned(16))) _Float16 VTs[64 * 64];     // 8 KB
    __shared__ __attribute__((aligned(16))) _Float16 Ps[4][16 * 64];   // 8 KB
    __shared__ __attribute__((aligned(16))) float    Bs[64 * 64];      // 16 KB

    const _Float16* qbase = qkv + (size_t)(b * NSEQ) * (3 * DIMC) + h * HD;
    const _Float16* kbase = qbase + DIMC;
    const _Float16* vbase = qbase + 2 * DIMC;

    // Q tile, pre-scaled by HD^-0.5 = 0.125 (exact in f16)
    #pragma unroll
    for (int c = 0; c < 4; ++c) {
        int chunk = tid + c * 128;       // 0..511
        int row = chunk >> 3;
        int cc = chunk & 7;
        HU8 u;
        u.v = *(const v8h*)&qbase[(size_t)(qb * 64 + row) * (3 * DIMC) + cc * 8];
        #pragma unroll
        for (int e = 0; e < 8; ++e) u.f[e] = u.f[e] * (_Float16)0.125f;
        *(v8h*)&Qs[row * 64 + cc * 8] = u.v;
    }
    __syncthreads();

    // per-wave Q A-fragments (rows w*16 .. w*16+15), two 32-wide K chunks
    const unsigned* Qu = (const unsigned*)Qs;
    U16 aQ[2];
    #pragma unroll
    for (int part = 0; part < 2; ++part) {
        int row = w * 16 + l15;
        #pragma unroll
        for (int v = 0; v < 8; ++v) {
            int kidx = part * 32 + ((v & 4) ? 16 : 0) + hi * 8 + (v & 3) * 2;
            aQ[part].u[v] = Qu[row * 32 + (kidx >> 1)];
        }
    }

    float mrow[8], lrow[8];
    #pragma unroll
    for (int r = 0; r < 8; ++r) { mrow[r] = -1e30f; lrow[r] = 0.f; }
    v8f oacc[4];
    #pragma unroll
    for (int t = 0; t < 4; ++t) oacc[t] = {};

    for (int kb = 0; kb < 16; ++kb) {
        __syncthreads();
        // K tile, row-major [key][hd]
        #pragma unroll
        for (int c = 0; c < 4; ++c) {
            int chunk = tid + c * 128;
            int row = chunk >> 3;
            int cc = chunk & 7;
            *(v8h*)&Ks[row * 64 + cc * 8] =
                *(const v8h*)&kbase[(size_t)(kb * 64 + row) * (3 * DIMC) + cc * 8];
        }
        // V tile, transposed into LDS: VTs[hd][key]
        {
            int key = tid >> 1;
            int hb = (tid & 1) * 32;
            HU8 u[4];
            #pragma unroll
            for (int c = 0; c < 4; ++c)
                u[c].v = *(const v8h*)&vbase[(size_t)(kb * 64 + key) * (3 * DIMC) + hb + c * 8];
            #pragma unroll
            for (int c = 0; c < 4; ++c)
                #pragma unroll
                for (int e = 0; e < 8; ++e)
                    VTs[(hb + c * 8 + e) * 64 + key] = u[c].f[e];
        }
        // combined additive tile: pbias + mask (+ lmask), staged as f32
        #pragma unroll
        for (int c = 0; c < 8; ++c) {
            int chunk = tid + c * 128;        // 0..1023 float4 chunks
            int row = chunk >> 4;             // 0..63
            int cc = (chunk & 15) * 4;        // 0..60
            int qr = qb * 64 + row;
            int kc = kb * 64 + cc;
            f32x4 v = *(const f32x4*)&pbias[((size_t)bh * NSEQ + qr) * NSEQ + kc];
            f32x4 mv = *(const f32x4*)&mask[((size_t)b * NSEQ + qr) * NSEQ + kc];
            v = v + mv;
            if (lmask) {
                f32x4 lv = *(const f32x4*)&lmask[((size_t)b * NSEQ + qr) * NSEQ + kc];
                v = v + lv;
            }
            *(f32x4*)&Bs[row * 64 + cc] = v;
        }
        __syncthreads();

        const unsigned* Ku = (const unsigned*)Ks;
        const unsigned* Vu = (const unsigned*)VTs;

        // S = Q @ K^T  : 4 col tiles of 16 keys, 2 WMMA each (hd 0..31, 32..63)
        v8f sacc[4];
        #pragma unroll
        for (int nt = 0; nt < 4; ++nt) {
            U16 bk0, bk1;
            int key = nt * 16 + l15;
            #pragma unroll
            for (int v = 0; v < 8; ++v) {
                bk0.u[v] = Ku[key * 32 + ((v & 4) ? 8 : 0) + hi * 4 + (v & 3)];
                bk1.u[v] = Ku[key * 32 + 16 + ((v & 4) ? 8 : 0) + hi * 4 + (v & 3)];
            }
            v8f s = {};
            s = __builtin_amdgcn_wmma_f32_16x16x32_f16(false, aQ[0].h, false, bk0.h,
                                                       (short)0, s, false, false);
            s = __builtin_amdgcn_wmma_f32_16x16x32_f16(false, aQ[1].h, false, bk1.h,
                                                       (short)0, s, false, false);
            sacc[nt] = s;
        }

        // + staged (bias+mask) from LDS
        #pragma unroll
        for (int nt = 0; nt < 4; ++nt)
            #pragma unroll
            for (int r = 0; r < 8; ++r)
                sacc[nt][r] += Bs[(w * 16 + r + 8 * hi) * 64 + nt * 16 + l15];

        // online softmax: block row-max
        float mblk[8];
        #pragma unroll
        for (int r = 0; r < 8; ++r) {
            float m = fmaxf(fmaxf(sacc[0][r], sacc[1][r]),
                            fmaxf(sacc[2][r], sacc[3][r]));
            #pragma unroll
            for (int off = 1; off < 16; off <<= 1)
                m = fmaxf(m, __shfl_xor(m, off, 32));
            mblk[r] = m;
        }
        #pragma unroll
        for (int r = 0; r < 8; ++r) {
            float mnew = fmaxf(mrow[r], mblk[r]);
            float corr = __expf(mrow[r] - mnew);
            mrow[r] = mnew;
            lrow[r] *= corr;
            #pragma unroll
            for (int t = 0; t < 4; ++t) oacc[t][r] *= corr;
        }
        #pragma unroll
        for (int nt = 0; nt < 4; ++nt)
            #pragma unroll
            for (int r = 0; r < 8; ++r)
                sacc[nt][r] = __expf(sacc[nt][r] - mrow[r]);
        #pragma unroll
        for (int r = 0; r < 8; ++r) {
            float ps = sacc[0][r] + sacc[1][r] + sacc[2][r] + sacc[3][r];
            #pragma unroll
            for (int off = 1; off < 16; off <<= 1)
                ps += __shfl_xor(ps, off, 32);
            lrow[r] += ps;
        }

        // P (C-layout) -> per-wave LDS -> A-layout fragments
        _Float16* Pw = Ps[w];
        #pragma unroll
        for (int nt = 0; nt < 4; ++nt)
            #pragma unroll
            for (int r = 0; r < 8; ++r)
                Pw[(r + 8 * hi) * 64 + nt * 16 + l15] = (_Float16)sacc[nt][r];

        const unsigned* Pu = (const unsigned*)Pw;
        U16 pa[2];
        #pragma unroll
        for (int part = 0; part < 2; ++part)
            #pragma unroll
            for (int v = 0; v < 8; ++v) {
                int kidx = part * 32 + ((v & 4) ? 16 : 0) + hi * 8 + (v & 3) * 2;
                pa[part].u[v] = Pu[l15 * 32 + (kidx >> 1)];
            }

        // O += P @ V : B operand from VTs[hd][key] (key pairs contiguous)
        #pragma unroll
        for (int t = 0; t < 4; ++t) {
            U16 bv0, bv1;
            int hd = t * 16 + l15;
            #pragma unroll
            for (int v = 0; v < 8; ++v) {
                bv0.u[v] = Vu[hd * 32 + ((v & 4) ? 8 : 0) + hi * 4 + (v & 3)];
                bv1.u[v] = Vu[hd * 32 + 16 + ((v & 4) ? 8 : 0) + hi * 4 + (v & 3)];
            }
            oacc[t] = __builtin_amdgcn_wmma_f32_16x16x32_f16(false, pa[0].h, false, bv0.h,
                                                             (short)0, oacc[t], false, false);
            oacc[t] = __builtin_amdgcn_wmma_f32_16x16x32_f16(false, pa[1].h, false, bv1.h,
                                                             (short)0, oacc[t], false, false);
        }
    }

    // normalize and write o[token, h*64 + hd] f16
    #pragma unroll
    for (int t = 0; t < 4; ++t)
        #pragma unroll
        for (int r = 0; r < 8; ++r) {
            int row = b * NSEQ + qb * 64 + w * 16 + r + 8 * hi;
            int col = h * HD + t * 16 + l15;
            out[(size_t)row * DIMC + col] = (_Float16)(oacc[t][r] / lrow[r]);
        }
}

// ---------------------------------------------------------------------------
// Host-side orchestration
// ---------------------------------------------------------------------------
extern "C" void kernel_launch(void* const* d_in, const int* in_sizes, int n_in,
                              void* d_out, int out_size, void* d_ws, size_t ws_size,
                              hipStream_t stream) {
    (void)in_sizes; (void)n_in; (void)out_size; (void)ws_size;

    const float* x         = (const float*)d_in[0];
    const float* attn_mask = (const float*)d_in[1];
    const float* dist      = (const float*)d_in[2];
    const int*   edge      = (const int*)d_in[3];
    const float* local_msk = (const float*)d_in[4];
    const float* emb       = (const float*)d_in[5];
    const float* de_ln_g   = (const float*)d_in[6];
    const float* de_ln_b   = (const float*)d_in[7];
    const float* de_w      = (const float*)d_in[8];
    const float* de_b      = (const float*)d_in[9];
    const float* lng[3] = { (const float*)d_in[10], (const float*)d_in[15], (const float*)d_in[20] };
    const float* lnb[3] = { (const float*)d_in[11], (const float*)d_in[16], (const float*)d_in[21] };
    const float* qkvw_f32[2]  = { (const float*)d_in[12], (const float*)d_in[17] };
    const float* projw_f32[2] = { (const float*)d_in[13], (const float*)d_in[18] };
    const float* projb[2]     = { (const float*)d_in[14], (const float*)d_in[19] };
    const float* fc1w_f32 = (const float*)d_in[22];
    const float* fc1b     = (const float*)d_in[23];
    const float* fc2w_f32 = (const float*)d_in[24];
    const float* fc2b     = (const float*)d_in[25];

    // --- carve workspace ---
    char* ws = (char*)d_ws;
    size_t off = 0;
    auto carve = [&](size_t bytes) -> void* {
        void* p = ws + off;
        off = (off + bytes + 255) & ~(size_t)255;
        return p;
    };
    float*    pbias  = (float*)carve((size_t)NB * NH * NSEQ * NSEQ * 4);   // 192 MB
    _Float16* qkvw16[2], *projw16[2];
    qkvw16[0]  = (_Float16*)carve((size_t)3 * DIMC * DIMC * 2);
    qkvw16[1]  = (_Float16*)carve((size_t)3 * DIMC * DIMC * 2);
    projw16[0] = (_Float16*)carve((size_t)DIMC * DIMC * 2);
    projw16[1] = (_Float16*)carve((size_t)DIMC * DIMC * 2);
    _Float16* fc1w16 = (_Float16*)carve((size_t)HID * DIMC * 2);
    _Float16* fc2w16 = (_Float16*)carve((size_t)DIMC * HID * 2);
    float*    xcur   = (float*)carve((size_t)TOK * DIMC * 4);
    _Float16* xln    = (_Float16*)carve((size_t)TOK * DIMC * 2);
    _Float16* qkvbuf = (_Float16*)carve((size_t)TOK * 3 * DIMC * 2);
    _Float16* obuf   = (_Float16*)carve((size_t)TOK * DIMC * 2);
    _Float16* hbuf   = (_Float16*)carve((size_t)TOK * HID * 2);

    // --- weight conversion f32 -> f16 ---
    auto conv = [&](const float* src, _Float16* dst, int n) {
        f32_to_f16_kernel<<<(n + 255) / 256, 256, 0, stream>>>(src, dst, n);
    };
    conv(qkvw_f32[0],  qkvw16[0],  3 * DIMC * DIMC);
    conv(qkvw_f32[1],  qkvw16[1],  3 * DIMC * DIMC);
    conv(projw_f32[0], projw16[0], DIMC * DIMC);
    conv(projw_f32[1], projw16[1], DIMC * DIMC);
    conv(fc1w_f32, fc1w16, HID * DIMC);
    conv(fc2w_f32, fc2w16, DIMC * HID);

    // --- pair bias ---
    bias_kernel<<<(NB * NSEQ * NSEQ) / 256, 256, 0, stream>>>(
        dist, edge, emb, de_ln_g, de_ln_b, de_w, de_b, pbias);

    // --- residual stream init (TDM copy path) ---
    residual_copy_kernel<<<TOK / 16, 256, 0, stream>>>(x, xcur);

    // --- two attention blocks ---
    for (int blk = 0; blk < 2; ++blk) {
        ln_kernel<<<TOK, 256, 0, stream>>>(xcur, lng[blk], lnb[blk], xln);
        gemm_f16_wmma<<<dim3(3 * DIMC / 128, TOK / 128), 256, 0, stream>>>(
            xln, qkvw16[blk], qkvbuf, nullptr, nullptr,
            TOK, 3 * DIMC, DIMC, EPI_F16);
        flash_attn_kernel<<<dim3(NSEQ / 64, NB * NH), 128, 0, stream>>>(
            qkvbuf, pbias, attn_mask, (blk == 0) ? local_msk : nullptr, obuf);
        gemm_f16_wmma<<<dim3(DIMC / 128, TOK / 128), 256, 0, stream>>>(
            obuf, projw16[blk], xcur, projb[blk], xcur,
            TOK, DIMC, DIMC, EPI_F32_BIAS_RES);
    }

    // --- FFN ---
    ln_kernel<<<TOK, 256, 0, stream>>>(xcur, lng[2], lnb[2], xln);
    gemm_f16_wmma<<<dim3(HID / 128, TOK / 128), 256, 0, stream>>>(
        xln, fc1w16, hbuf, fc1b, nullptr,
        TOK, HID, DIMC, EPI_F16_BIAS_GELU);
    gemm_f16_wmma<<<dim3(DIMC / 128, TOK / 128), 256, 0, stream>>>(
        hbuf, fc2w16, (float*)d_out, fc2b, xcur,
        TOK, DIMC, HID, EPI_F32_BIAS_RES);
}
